// DensityModel_55568286875773
// MI455X (gfx1250) — matrix operand order repaired and checked by reading, not compiled, not run on hardware
//
#include <hip/hip_runtime.h>

// ---------------- static problem config (matches reference) ----------------
#define HDIM 128
#define BATCH 4
#define NATOM 64
#define EA_PER 2560
#define NPROBE_PER 4000
#define EP_PER 60000
#define NLAYERS 3
#define EDGE_FEAT 50

#define TOTAL_NODES (BATCH * NATOM)       // 256
#define TOTAL_PROBES (BATCH * NPROBE_PER) // 16000
#define TOTAL_AE (BATCH * EA_PER)         // 10240
#define TOTAL_PE (BATCH * EP_PER)         // 240000

// Per-wave LDS block stride: always 4 K-chunks (K=128) regardless of how many
// chunks a particular stage uses -> wave blocks never alias across stages.
#define WCH 4
#define CHUNK_HALVES (32 * 16)               // one K-chunk of a frag-A tile per wave
#define WAVE_BLOCK (WCH * CHUNK_HALVES)      // 2048 halves per wave
#define PACK_CHUNK (8 * 32 * 16)             // 4096 halves per packed-W K-chunk (8 col tiles)

typedef __attribute__((ext_vector_type(16))) _Float16 v16h;
typedef __attribute__((ext_vector_type(4)))  _Float16 h4;
typedef __attribute__((ext_vector_type(8)))  float    v8f;

// shifted softplus: softplus(x) - ln2, via 2 TRANS ops (exp+log) that co-issue
// with VALU/WMMA on CDNA5 (XDL WMMA is tracked as TRANS).
__device__ __forceinline__ float sspf(float v) {
  const float z = __expf(-fabsf(v));
  return fmaxf(v, 0.0f) + __logf(1.0f + z) - 0.69314718056f;
}
__device__ __forceinline__ float sigm(float v) {
  const float e = __expf(-v);
#if __has_builtin(__builtin_amdgcn_rcpf)
  return __builtin_amdgcn_rcpf(1.0f + e);
#else
  return 1.0f / (1.0f + e);
#endif
}
__device__ __forceinline__ void vclear(v8f (&a)[8]) {
#pragma unroll
  for (int t = 0; t < 8; ++t)
#pragma unroll
    for (int i = 0; i < 8; ++i) a[t][i] = 0.0f;
}

// ---- fragment-order index helpers (CDNA5 16x16x32 f16 layouts, wave32) ----
// A frag layout per K-chunk: [lane 0..31][h 0..15] contiguous; element (m, kk):
//   lane = m | (((kk>>3)&1)<<4),  h = (kk&7) | ((kk>>4)<<3)
// B frag layout per K-chunk/tile: [lane][h]; element (kk, n):
//   lane = n | (((kk>>3)&1)<<4),  h = (kk&7) | ((kk>>4)<<3)

// GEMM: C[64x128] += A(frag LDS, this wave's 16 rows) @ W(frag-packed f16, global)
template <int NCHUNKS>
__device__ __forceinline__ void gemm_frag(v8f (&acc)[8], const _Float16* Afrag,
                                          const _Float16* Wf) {
  const int wave = threadIdx.x >> 5, lane = threadIdx.x & 31;
  const _Float16* Aw = Afrag + wave * WAVE_BLOCK + (lane << 4);
#pragma unroll
  for (int c = 0; c < NCHUNKS; ++c) {
    const v16h a = *(const v16h*)(Aw + c * CHUNK_HALVES);
#pragma unroll
    for (int t = 0; t < 8; ++t) {
      const v16h b = *(const v16h*)(Wf + (((c << 3) + t) << 9) + (lane << 4));
      acc[t] = __builtin_amdgcn_wmma_f32_16x16x32_f16(
          false, a, false, b, (short)0, acc[t], false, false);
    }
  }
}

// Stage this wave's 16 rows of a (possibly gathered) f32 feature matrix [*,128]
// into the wave's frag-A LDS block (K = 128, 4 chunks).
__device__ __forceinline__ void stage_frag_rows(_Float16* Afrag, const float* feat,
                                                const int* gidx, int row0, int clampN) {
  const int wave = threadIdx.x >> 5, lane = threadIdx.x & 31;
  _Float16* Aw = Afrag + wave * WAVE_BLOCK;
  for (int q = lane; q < 16 * 32; q += 32) {  // 16 rows x 32 quads (128/4)
    const int rl = (wave << 4) + (q >> 5);
    const int c = (q & 31) << 2;
    int src = min(row0 + rl, clampN - 1);
    if (gidx) src = gidx[src];
    const float4 v = *(const float4*)(feat + (size_t)src * HDIM + c);
    h4 hv;
    hv[0] = (_Float16)v.x; hv[1] = (_Float16)v.y;
    hv[2] = (_Float16)v.z; hv[3] = (_Float16)v.w;
    const int chunk = c >> 5, kk = c & 31;
    const int h = (kk & 7) | ((kk >> 4) << 3);
    const int dl = (rl & 15) | (((kk >> 3) & 1) << 4);
    *(h4*)(Aw + chunk * CHUNK_HALVES + (dl << 4) + h) = hv;
  }
}

// Epilogue: Hfrag (wave-private, K=128 frag-A order) = f16(ssp(acc + bias))
__device__ __forceinline__ void store_act_ssp_frag(const v8f (&acc)[8], _Float16* Hfrag,
                                                   const float* bias) {
  const int wave = threadIdx.x >> 5, lane = threadIdx.x & 31;
  const int mrow = (lane >> 4) << 3, ncol = lane & 15;
  _Float16* Hw = Hfrag + wave * WAVE_BLOCK;
#pragma unroll
  for (int t = 0; t < 8; ++t) {
    const int col = (t << 4) + ncol;
    const float b = bias[col];
    const int chunk = col >> 5, kk = col & 31;
    const int h = (kk & 7) | ((kk >> 4) << 3);
    const int lh = ((kk >> 3) & 1) << 4;
#pragma unroll
    for (int r = 0; r < 8; ++r) {
      const int m = mrow + r;
      Hw[chunk * CHUNK_HALVES + ((m | lh) << 4) + h] = (_Float16)sspf(acc[t][r] + b);
    }
  }
}

// ---------------- weight packing: f32 [L,Kreal,128] -> f16 frag-B order ----------------
// out layout per layer: [Kpad/32][8 col tiles][32 lanes][16 h]
__global__ void pack_weight_kernel(const float* __restrict__ W, _Float16* __restrict__ out,
                                   int L, int Kreal, int Kpad) {
  const int nchunks = Kpad >> 5;
  const int total = L * nchunks * 8 * 32;
  int gid = blockIdx.x * blockDim.x + threadIdx.x;
  if (gid >= total) return;
  const int lane = gid & 31;
  const int t = (gid >> 5) & 7;
  const int chunk = (gid >> 8) % nchunks;
  const int l = gid / (nchunks << 8);
  const int n = lane & 15;
  const int kh = (lane >> 4) << 3;
  const float* Wl = W + (size_t)l * Kreal * HDIM;
  _Float16* o = out + (size_t)l * nchunks * PACK_CHUNK + ((((chunk << 3) + t) << 5) + lane) * 16;
#pragma unroll
  for (int h = 0; h < 16; ++h) {
    const int kg = (chunk << 5) + ((h & 8) << 1) + kh + (h & 7);
    o[h] = (_Float16)((kg < Kreal) ? Wl[kg * HDIM + (t << 4) + n] : 0.0f);
  }
}

// ---------------- small prep kernels ----------------
__global__ void prep_edges_kernel(const int* __restrict__ edges,
                                  const int* __restrict__ num_nodes,
                                  const int* __restrict__ num_probes,
                                  int Eper, int recv_probe,
                                  int* __restrict__ send, int* __restrict__ recv, int Etot) {
  int i = blockIdx.x * blockDim.x + threadIdx.x;
  if (i >= Etot) return;
  int b = i / Eper;
  int noff = 0, poff = 0;
  for (int j = 0; j < b; ++j) { noff += num_nodes[j]; poff += num_probes[j]; }
  send[i] = edges[i * 2 + 0] + noff;
  recv[i] = edges[i * 2 + 1] + (recv_probe ? poff : noff);
}

__global__ void embed_kernel(const int* __restrict__ nodes, const float* __restrict__ emb,
                             float* __restrict__ x, int n) {
  int i = blockIdx.x * blockDim.x + threadIdx.x;
  if (i < n) x[i] = emb[nodes[i >> 7] * HDIM + (i & 127)];
}

__global__ void zero_kernel(float* __restrict__ p, int n) {
  int i = blockIdx.x * blockDim.x + threadIdx.x;
  if (i < n) p[i] = 0.0f;
}

// ---------------- SchnetMessageFunction + scatter-sum (atoms & probes) ----------------
// 128 threads = 4 independent wave pipelines; no barriers (wave-private LDS blocks).
__global__ __launch_bounds__(128)
void edge_message_kernel(int E, const float* __restrict__ dist,
                         const int* __restrict__ send, const int* __restrict__ recv,
                         const float* __restrict__ sfeat, const float* __restrict__ rfeat,
                         const _Float16* __restrict__ ew1p, const float* __restrict__ eb1,
                         const _Float16* __restrict__ ew2p, const float* __restrict__ eb2,
                         const _Float16* __restrict__ nw1p, const float* __restrict__ nb1,
                         const _Float16* __restrict__ nw2p, const float* __restrict__ nb2,
                         float* __restrict__ msum) {
  extern __shared__ char smem[];
  _Float16* Abuf  = (_Float16*)smem;            // 4 waves * 2048 halves
  _Float16* Hfrag = Abuf + 4 * WAVE_BLOCK;      // 4 waves * 2048 halves
  float*    cut   = (float*)(Hfrag + 4 * WAVE_BLOCK); // 64 floats (wave-private slices)
  const int e0 = blockIdx.x * 64;
  const int wave = threadIdx.x >> 5, lane = threadIdx.x & 31;
  const int mrow = (lane >> 4) << 3, ncol = lane & 15;

  // --- stage gaussian-expanded edge features (K padded 50->64, 2 chunks) ---
  {
    _Float16* Aw = Abuf + wave * WAVE_BLOCK;
    for (int q = lane; q < 16 * 16; q += 32) {  // 16 rows x 16 quads (64/4)
      const int rl = (wave << 4) + (q >> 4);
      const int c = (q & 15) << 2;
      const int e = min(e0 + rl, E - 1);
      const float d = dist[e];
      h4 hv;
#pragma unroll
      for (int j = 0; j < 4; ++j) {
        const int cc = c + j;
        float v = 0.0f;
        if (cc < EDGE_FEAT) {
          const float dd = d - 0.1f * (float)cc;
          v = __expf(-dd * dd * 50.0f); // 1/(2*step^2) = 50
        }
        hv[j] = (_Float16)v;
      }
      const int chunk = c >> 5, kk = c & 31;
      const int h = (kk & 7) | ((kk >> 4) << 3);
      const int dl = (rl & 15) | (((kk >> 3) & 1) << 4);
      *(h4*)(Aw + chunk * CHUNK_HALVES + (dl << 4) + h) = hv;
    }
    if (lane < 16) {
      const int rl = (wave << 4) + lane;
      const int e = min(e0 + rl, E - 1);
      cut[rl] = 1.0f - sigm(5.0f * (dist[e] - 3.5f)); // cutoff - 1.5
    }
  }

  v8f acc[8], gate[8];

  // edge MLP layer 1: hidden = ssp(F @ ew1 + eb1)
  vclear(acc);
  gemm_frag<2>(acc, Abuf, ew1p);
  store_act_ssp_frag(acc, Hfrag, eb1);

  // edge MLP layer 2 -> gate (registers), * soft cutoff
  vclear(gate);
  gemm_frag<4>(gate, Hfrag, ew2p);
#pragma unroll
  for (int t = 0; t < 8; ++t) {
    const float b = eb2[(t << 4) + ncol];
#pragma unroll
    for (int r = 0; r < 8; ++r) {
      const int row = (wave << 4) + mrow + r;
      gate[t][r] = (gate[t][r] + b) * cut[row];
    }
  }

  // node MLP layer 1 on concat(sender, receiver): two K-halves through Abuf
  stage_frag_rows(Abuf, sfeat, send, e0, E);
  vclear(acc);
  gemm_frag<4>(acc, Abuf, nw1p);                       // K rows 0..127
  stage_frag_rows(Abuf, rfeat, recv, e0, E);
  gemm_frag<4>(acc, Abuf, nw1p + 4 * PACK_CHUNK);      // K rows 128..255
  store_act_ssp_frag(acc, Hfrag, nb1);

  // node MLP layer 2, gate, scatter-sum to receivers
  vclear(acc);
  gemm_frag<4>(acc, Hfrag, nw2p);
#pragma unroll
  for (int t = 0; t < 8; ++t) {
    const int col = (t << 4) + ncol;
    const float b = nb2[col];
#pragma unroll
    for (int r = 0; r < 8; ++r) {
      const int row = (wave << 4) + mrow + r;
      const int e = e0 + row;
      if (e < E) {
        const float m = (acc[t][r] + b) * gate[t][r];
        atomicAdd(&msum[recv[e] * HDIM + col], m);
      }
    }
  }
}

// ---------------- probe state update: ps = ps*g + (1-g)*mlp2(msum) ----------------
__global__ __launch_bounds__(128)
void probe_update_kernel(float* __restrict__ ps, const float* __restrict__ msum,
                         const _Float16* __restrict__ gw1p, const float* __restrict__ gb1,
                         const _Float16* __restrict__ gw2p, const float* __restrict__ gb2,
                         const _Float16* __restrict__ tw1p, const float* __restrict__ tb1,
                         const _Float16* __restrict__ tw2p, const float* __restrict__ tb2,
                         int NPtot) {
  extern __shared__ char smem[];
  _Float16* Abuf  = (_Float16*)smem;
  _Float16* Hfrag = Abuf + 4 * WAVE_BLOCK;
  const int p0 = blockIdx.x * 64;
  const int wave = threadIdx.x >> 5, lane = threadIdx.x & 31;
  const int mrow = (lane >> 4) << 3, ncol = lane & 15;

  v8f acc[8], g[8];

  stage_frag_rows(Abuf, ps, nullptr, p0, NPtot);
  vclear(acc);
  gemm_frag<4>(acc, Abuf, gw1p);
  store_act_ssp_frag(acc, Hfrag, gb1);
  vclear(g);
  gemm_frag<4>(g, Hfrag, gw2p);
#pragma unroll
  for (int t = 0; t < 8; ++t) {
    const float b = gb2[(t << 4) + ncol];
#pragma unroll
    for (int r = 0; r < 8; ++r) g[t][r] = sigm(g[t][r] + b);
  }

  stage_frag_rows(Abuf, msum, nullptr, p0, NPtot);
  vclear(acc);
  gemm_frag<4>(acc, Abuf, tw1p);
  store_act_ssp_frag(acc, Hfrag, tb1);
  vclear(acc);
  gemm_frag<4>(acc, Hfrag, tw2p);
#pragma unroll
  for (int t = 0; t < 8; ++t) {
    const int col = (t << 4) + ncol;
    const float b = tb2[col];
#pragma unroll
    for (int r = 0; r < 8; ++r) {
      const int row = (wave << 4) + mrow + r;
      const int p = p0 + row;
      if (p < NPtot) {
        const float old = ps[p * HDIM + col];
        const float gv = g[t][r];
        ps[p * HDIM + col] = old * gv + (1.0f - gv) * (acc[t][r] + b);
      }
    }
  }
}

// ---------------- atom state update: x += mlp2(msum); rep = x ----------------
__global__ __launch_bounds__(128)
void atom_update_kernel(float* __restrict__ x, float* __restrict__ rep,
                        const float* __restrict__ msum,
                        const _Float16* __restrict__ w1p, const float* __restrict__ b1,
                        const _Float16* __restrict__ w2p, const float* __restrict__ b2,
                        int Ntot) {
  extern __shared__ char smem[];
  _Float16* Abuf  = (_Float16*)smem;
  _Float16* Hfrag = Abuf + 4 * WAVE_BLOCK;
  const int n0 = blockIdx.x * 64;
  const int wave = threadIdx.x >> 5, lane = threadIdx.x & 31;
  const int mrow = (lane >> 4) << 3, ncol = lane & 15;

  v8f acc[8];
  stage_frag_rows(Abuf, msum, nullptr, n0, Ntot);
  vclear(acc);
  gemm_frag<4>(acc, Abuf, w1p);
  store_act_ssp_frag(acc, Hfrag, b1);
  vclear(acc);
  gemm_frag<4>(acc, Hfrag, w2p);
#pragma unroll
  for (int t = 0; t < 8; ++t) {
    const int col = (t << 4) + ncol;
    const float b = b2[col];
#pragma unroll
    for (int r = 0; r < 8; ++r) {
      const int row = (wave << 4) + mrow + r;
      const int n = n0 + row;
      if (n < Ntot) {
        const float xv = x[n * HDIM + col] + acc[t][r] + b;
        x[n * HDIM + col] = xv;
        rep[n * HDIM + col] = xv;
      }
    }
  }
}

// ---------------- readout: out = ssp(ps@w1+b1)@w2 + b2 ----------------
__global__ __launch_bounds__(128)
void readout_kernel(const float* __restrict__ ps,
                    const _Float16* __restrict__ w1p, const float* __restrict__ b1,
                    const float* __restrict__ w2, const float* __restrict__ b2,
                    float* __restrict__ out, int NPtot) {
  extern __shared__ char smem[];
  _Float16* Abuf = (_Float16*)smem; // 4 waves * 2048 halves
  const int p0 = blockIdx.x * 64;
  const int wave = threadIdx.x >> 5, lane = threadIdx.x & 31;
  const int mrow = (lane >> 4) << 3, ncol = lane & 15;

  v8f acc[8];
  stage_frag_rows(Abuf, ps, nullptr, p0, NPtot);
  vclear(acc);
  gemm_frag<4>(acc, Abuf, w1p);

  float partial[8];
#pragma unroll
  for (int r = 0; r < 8; ++r) partial[r] = 0.0f;
#pragma unroll
  for (int t = 0; t < 8; ++t) {
    const int col = (t << 4) + ncol;
    const float b = b1[col];
    const float wv = w2[col]; // w2 is [128,1]
#pragma unroll
    for (int r = 0; r < 8; ++r) partial[r] += sspf(acc[t][r] + b) * wv;
  }
  // reduce across the 16 lanes of each half-wave (wave32 shuffles)
#pragma unroll
  for (int r = 0; r < 8; ++r) {
    float v = partial[r];
    v += __shfl_xor(v, 1);
    v += __shfl_xor(v, 2);
    v += __shfl_xor(v, 4);
    v += __shfl_xor(v, 8);
    if (ncol == 0) {
      const int p = p0 + (wave << 4) + mrow + r;
      if (p < NPtot) out[p] = v + b2[0];
    }
  }
}

// ---------------- host orchestration ----------------
extern "C" void kernel_launch(void* const* d_in, const int* in_sizes, int n_in,
                              void* d_out, int out_size, void* d_ws, size_t ws_size,
                              hipStream_t stream) {
  (void)in_sizes; (void)n_in; (void)out_size; (void)ws_size;
  // positional inputs (setup_inputs dict order)
  const int*   nodes      = (const int*)d_in[0];
  const int*   a_edges    = (const int*)d_in[1];
  const float* a_dist     = (const float*)d_in[2];
  const int*   num_nodes  = (const int*)d_in[3];
  const int*   p_edges    = (const int*)d_in[5];
  const float* p_dist     = (const float*)d_in[6];
  const int*   num_probes = (const int*)d_in[7];
  // params flattened as a JAX pytree: dict keys in sorted order
  const float* amsg_edge_b1 = (const float*)d_in[9];
  const float* amsg_edge_b2 = (const float*)d_in[10];
  const float* amsg_edge_w1 = (const float*)d_in[11];
  const float* amsg_edge_w2 = (const float*)d_in[12];
  const float* amsg_node_b1 = (const float*)d_in[13];
  const float* amsg_node_b2 = (const float*)d_in[14];
  const float* amsg_node_w1 = (const float*)d_in[15];
  const float* amsg_node_w2 = (const float*)d_in[16];
  const float* atom_embed   = (const float*)d_in[17];
  const float* atrans_b1    = (const float*)d_in[18];
  const float* atrans_b2    = (const float*)d_in[19];
  const float* atrans_w1    = (const float*)d_in[20];
  const float* atrans_w2    = (const float*)d_in[21];
  const float* pgate_b1     = (const float*)d_in[22];
  const float* pgate_b2     = (const float*)d_in[23];
  const float* pgate_w1     = (const float*)d_in[24];
  const float* pgate_w2     = (const float*)d_in[25];
  const float* pmsg_edge_b1 = (const float*)d_in[26];
  const float* pmsg_edge_b2 = (const float*)d_in[27];
  const float* pmsg_edge_w1 = (const float*)d_in[28];
  const float* pmsg_edge_w2 = (const float*)d_in[29];
  const float* pmsg_node_b1 = (const float*)d_in[30];
  const float* pmsg_node_b2 = (const float*)d_in[31];
  const float* pmsg_node_w1 = (const float*)d_in[32];
  const float* pmsg_node_w2 = (const float*)d_in[33];
  const float* ptrans_b1    = (const float*)d_in[34];
  const float* ptrans_b2    = (const float*)d_in[35];
  const float* ptrans_w1    = (const float*)d_in[36];
  const float* ptrans_w2    = (const float*)d_in[37];
  const float* readout_b1   = (const float*)d_in[38];
  const float* readout_b2   = (const float*)d_in[39];
  const float* readout_w1   = (const float*)d_in[40];
  const float* readout_w2   = (const float*)d_in[41];

  // workspace carve (~21 MB)
  char* w = (char*)d_ws;
  auto take = [&](size_t bytes) {
    char* p = w;
    w += (bytes + 255) & ~(size_t)255;
    return p;
  };
  float* x    = (float*)take((size_t)TOTAL_NODES * HDIM * 4);
  float* reps = (float*)take((size_t)NLAYERS * TOTAL_NODES * HDIM * 4);
  float* msA  = (float*)take((size_t)TOTAL_NODES * HDIM * 4);
  float* ps   = (float*)take((size_t)TOTAL_PROBES * HDIM * 4);
  float* msP  = (float*)take((size_t)TOTAL_PROBES * HDIM * 4);
  int* aS = (int*)take((size_t)TOTAL_AE * 4);
  int* aR = (int*)take((size_t)TOTAL_AE * 4);
  int* pS = (int*)take((size_t)TOTAL_PE * 4);
  int* pR = (int*)take((size_t)TOTAL_PE * 4);

  auto takeh = [&](size_t halves) { return (_Float16*)take(halves * 2); };
  // packed weights: per layer size = (Kpad/32) * PACK_CHUNK halves
  _Float16* aew1 = takeh((size_t)NLAYERS * 2 * PACK_CHUNK);
  _Float16* aew2 = takeh((size_t)NLAYERS * 4 * PACK_CHUNK);
  _Float16* anw1 = takeh((size_t)NLAYERS * 8 * PACK_CHUNK);
  _Float16* anw2 = takeh((size_t)NLAYERS * 4 * PACK_CHUNK);
  _Float16* atw1 = takeh((size_t)NLAYERS * 4 * PACK_CHUNK);
  _Float16* atw2 = takeh((size_t)NLAYERS * 4 * PACK_CHUNK);
  _Float16* pew1 = takeh((size_t)NLAYERS * 2 * PACK_CHUNK);
  _Float16* pew2 = takeh((size_t)NLAYERS * 4 * PACK_CHUNK);
  _Float16* pnw1 = takeh((size_t)NLAYERS * 8 * PACK_CHUNK);
  _Float16* pnw2 = takeh((size_t)NLAYERS * 4 * PACK_CHUNK);
  _Float16* pgw1 = takeh((size_t)NLAYERS * 4 * PACK_CHUNK);
  _Float16* pgw2 = takeh((size_t)NLAYERS * 4 * PACK_CHUNK);
  _Float16* ptw1 = takeh((size_t)NLAYERS * 4 * PACK_CHUNK);
  _Float16* ptw2 = takeh((size_t)NLAYERS * 4 * PACK_CHUNK);
  _Float16* row1 = takeh((size_t)1 * 4 * PACK_CHUNK);

  auto pack = [&](const float* W, _Float16* outp, int L, int Kreal, int Kpad) {
    const int total = L * (Kpad >> 5) * 256;
    pack_weight_kernel<<<(total + 127) / 128, 128, 0, stream>>>(W, outp, L, Kreal, Kpad);
  };
  pack(amsg_edge_w1, aew1, NLAYERS, EDGE_FEAT, 64);
  pack(amsg_edge_w2, aew2, NLAYERS, HDIM, HDIM);
  pack(amsg_node_w1, anw1, NLAYERS, 2 * HDIM, 2 * HDIM);
  pack(amsg_node_w2, anw2, NLAYERS, HDIM, HDIM);
  pack(atrans_w1,    atw1, NLAYERS, HDIM, HDIM);
  pack(atrans_w2,    atw2, NLAYERS, HDIM, HDIM);
  pack(pmsg_edge_w1, pew1, NLAYERS, EDGE_FEAT, 64);
  pack(pmsg_edge_w2, pew2, NLAYERS, HDIM, HDIM);
  pack(pmsg_node_w1, pnw1, NLAYERS, 2 * HDIM, 2 * HDIM);
  pack(pmsg_node_w2, pnw2, NLAYERS, HDIM, HDIM);
  pack(pgate_w1,     pgw1, NLAYERS, HDIM, HDIM);
  pack(pgate_w2,     pgw2, NLAYERS, HDIM, HDIM);
  pack(ptrans_w1,    ptw1, NLAYERS, HDIM, HDIM);
  pack(ptrans_w2,    ptw2, NLAYERS, HDIM, HDIM);
  pack(readout_w1,   row1, 1, HDIM, HDIM);

  const size_t lds_edge = (size_t)(8 * WAVE_BLOCK) * 2 + 64 * 4; // A + Hid + cut
  const size_t lds_upd  = (size_t)(8 * WAVE_BLOCK) * 2;          // A + Hid
  const size_t lds_ro   = (size_t)(4 * WAVE_BLOCK) * 2;          // A only

  prep_edges_kernel<<<(TOTAL_AE + 255) / 256, 256, 0, stream>>>(
      a_edges, num_nodes, num_probes, EA_PER, 0, aS, aR, TOTAL_AE);
  prep_edges_kernel<<<(TOTAL_PE + 255) / 256, 256, 0, stream>>>(
      p_edges, num_nodes, num_probes, EP_PER, 1, pS, pR, TOTAL_PE);
  embed_kernel<<<(TOTAL_NODES * HDIM + 255) / 256, 256, 0, stream>>>(
      nodes, atom_embed, x, TOTAL_NODES * HDIM);
  zero_kernel<<<(TOTAL_PROBES * HDIM + 255) / 256, 256, 0, stream>>>(
      ps, TOTAL_PROBES * HDIM);

  // ---- atom representation layers ----
  for (int l = 0; l < NLAYERS; ++l) {
    zero_kernel<<<(TOTAL_NODES * HDIM + 255) / 256, 256, 0, stream>>>(msA, TOTAL_NODES * HDIM);
    edge_message_kernel<<<TOTAL_AE / 64, 128, lds_edge, stream>>>(
        TOTAL_AE, a_dist, aS, aR, x, x,
        aew1 + (size_t)l * 2 * PACK_CHUNK, amsg_edge_b1 + (size_t)l * HDIM,
        aew2 + (size_t)l * 4 * PACK_CHUNK, amsg_edge_b2 + (size_t)l * HDIM,
        anw1 + (size_t)l * 8 * PACK_CHUNK, amsg_node_b1 + (size_t)l * HDIM,
        anw2 + (size_t)l * 4 * PACK_CHUNK, amsg_node_b2 + (size_t)l * HDIM,
        msA);
    atom_update_kernel<<<TOTAL_NODES / 64, 128, lds_upd, stream>>>(
        x, reps + (size_t)l * TOTAL_NODES * HDIM, msA,
        atw1 + (size_t)l * 4 * PACK_CHUNK, atrans_b1 + (size_t)l * HDIM,
        atw2 + (size_t)l * 4 * PACK_CHUNK, atrans_b2 + (size_t)l * HDIM,
        TOTAL_NODES);
  }

  // ---- probe message layers ----
  for (int l = 0; l < NLAYERS; ++l) {
    zero_kernel<<<(TOTAL_PROBES * HDIM + 255) / 256, 256, 0, stream>>>(msP, TOTAL_PROBES * HDIM);
    edge_message_kernel<<<TOTAL_PE / 64, 128, lds_edge, stream>>>(
        TOTAL_PE, p_dist, pS, pR, reps + (size_t)l * TOTAL_NODES * HDIM, ps,
        pew1 + (size_t)l * 2 * PACK_CHUNK, pmsg_edge_b1 + (size_t)l * HDIM,
        pew2 + (size_t)l * 4 * PACK_CHUNK, pmsg_edge_b2 + (size_t)l * HDIM,
        pnw1 + (size_t)l * 8 * PACK_CHUNK, pmsg_node_b1 + (size_t)l * HDIM,
        pnw2 + (size_t)l * 4 * PACK_CHUNK, pmsg_node_b2 + (size_t)l * HDIM,
        msP);
    probe_update_kernel<<<TOTAL_PROBES / 64, 128, lds_upd, stream>>>(
        ps, msP,
        pgw1 + (size_t)l * 4 * PACK_CHUNK, pgate_b1 + (size_t)l * HDIM,
        pgw2 + (size_t)l * 4 * PACK_CHUNK, pgate_b2 + (size_t)l * HDIM,
        ptw1 + (size_t)l * 4 * PACK_CHUNK, ptrans_b1 + (size_t)l * HDIM,
        ptw2 + (size_t)l * 4 * PACK_CHUNK, ptrans_b2 + (size_t)l * HDIM,
        TOTAL_PROBES);
  }

  readout_kernel<<<TOTAL_PROBES / 64, 128, lds_ro, stream>>>(
      ps, row1, readout_b1, readout_w2, readout_b2, (float*)d_out, TOTAL_PROBES);
}
